// FlatCritic_22230750724345
// MI455X (gfx1250) — matrix operand order repaired and checked by reading, not compiled, not run on hardware
//
#include <hip/hip_runtime.h>
#include <hip/hip_bf16.h>

typedef __attribute__((ext_vector_type(16))) __bf16 v16bf;
typedef __attribute__((ext_vector_type(8)))  __bf16 v8bf;
typedef __attribute__((ext_vector_type(8)))  float  v8f;

#define BB 128
#define LL 512
#define DDESC 768
#define DSTD 128
#define DACT 128
#define HH 512
#define DCTX 1152

// ---- workspace layout (bytes) ----
#define OFF_WDT   0
#define OFF_WST   196608
#define OFF_W1ET  229376
#define OFF_W2T   360448
#define OFF_CTXP  884736

// -------------------- prologue 1: transpose+convert weights to bf16 [N][K] -----------
__global__ void prep_weights(const float* __restrict__ Wd, const float* __restrict__ Ws,
                             const float* __restrict__ W1, const float* __restrict__ W2,
                             __bf16* __restrict__ wdT, __bf16* __restrict__ wsT,
                             __bf16* __restrict__ w1eT, __bf16* __restrict__ w2T) {
    const int total = 98304 + 16384 + 65536 + 262144;
    for (int i = blockIdx.x * blockDim.x + threadIdx.x; i < total; i += gridDim.x * blockDim.x) {
        int j = i;
        if (j < 98304) {                   // W_desc [768,128] -> wdT [128][768]
            int n = j / DDESC, k = j % DDESC;
            wdT[j] = (__bf16)Wd[k * DACT + n];
        } else if ((j -= 98304) < 16384) { // W_std [128,128] -> wsT [128][128]
            int n = j / DSTD, k = j % DSTD;
            wsT[j] = (__bf16)Ws[k * DACT + n];
        } else if ((j -= 16384) < 65536) { // W1 rows 1152..1279 -> w1eT [512][128]
            int n = j / DACT, k = j % DACT;
            w1eT[j] = (__bf16)W1[(DCTX + k) * HH + n];
        } else {                           // W2 [512,512] -> w2T [512][512]
            j -= 65536;
            int n = j / HH, k = j % HH;
            w2T[j] = (__bf16)W2[k * HH + n];
        }
    }
}

// -------------------- prologue 2: ctx part of first layer (per-b, reused across L) ---
__global__ void ctx_gemm(const float* __restrict__ instr, const float* __restrict__ st,
                         const float* __restrict__ hid, const float* __restrict__ W1,
                         const float* __restrict__ b1, float* __restrict__ ctxp) {
    int i = blockIdx.x * blockDim.x + threadIdx.x;
    if (i >= BB * HH) return;
    int b = i >> 9, h = i & (HH - 1);
    float acc = b1[h];
    for (int d = 0; d < 512; ++d) acc += instr[b * 512 + d] * W1[d * HH + h];
    for (int d = 0; d < 512; ++d) acc += st[b * 512 + d] * W1[(512 + d) * HH + h];
    for (int d = 0; d < 128; ++d) acc += hid[b * 128 + d] * W1[(1024 + d) * HH + h];
    ctxp[i] = acc;
}

// -------------------- WMMA fragment loads (contiguous k order, same for A and B) ---------------
__device__ __forceinline__ v16bf load_a_frag(const __bf16* base, int stride, int r0, int kc, int lane) {
    // 16x32 bf16 A: lanes 0-15 -> K 0..15; lanes 16-31 -> K 16..31
    int m  = r0 + (lane & 15);
    int kb = (lane & 16) ? 16 : 0;
    const __bf16* p = base + m * stride + kc + kb;
    union { v16bf v; v8bf h[2]; } u;
    u.h[0] = *(const v8bf*)(p);
    u.h[1] = *(const v8bf*)(p + 8);
    return u.v;
}
__device__ __forceinline__ v16bf load_b_frag(const __bf16* bt, int stride, int n0, int lane) {
    // 32x16 bf16 B from [N][K] LDS: lanes 0-15 -> K 0..15; lanes 16-31 -> K 16..31
    int n  = n0 + (lane & 15);
    int kb = (lane & 16) ? 16 : 0;
    const __bf16* p = bt + n * stride + kb;
    union { v16bf v; v8bf h[2]; } u;
    u.h[0] = *(const v8bf*)(p);
    u.h[1] = *(const v8bf*)(p + 8);
    return u.v;
}
#define WMMA_BF16(A, Bf, C) __builtin_amdgcn_wmma_f32_16x16x32_bf16(false, (A), false, (Bf), (short)0, (C), false, false)

// -------------------- CDNA5 async global->LDS (ASYNCcnt path) ------------------------
__device__ __forceinline__ void async_b128(unsigned ldsoff, const __bf16* base, unsigned gbyteoff) {
    asm volatile("global_load_async_to_lds_b128 %0, %1, %2"
                 :: "v"(ldsoff), "v"(gbyteoff), "s"(base) : "memory");
}
__device__ __forceinline__ void wait_async() {
#if __has_builtin(__builtin_amdgcn_s_wait_asynccnt)
    __builtin_amdgcn_s_wait_asynccnt(0);
#else
    asm volatile("s_wait_asynccnt 0x0" ::: "memory");
#endif
}

// -------------------- main fused kernel: 64 L-rows per WG, 256 threads ---------------
// LDS carve (byte offsets): sEmb 64x136 bf16, sH1 64x520 bf16,
//                           sA[2] 64x40 bf16, sBt[2] 512x40 bf16, qbuf 64 f32
#define SEMB_STRIDE 136
#define SH1_STRIDE  520
#define SA_STRIDE   40
#define SBT_STRIDE  40
#define LOFF_EMB    0
#define LOFF_H1     (64 * SEMB_STRIDE * 2)            // 17408
#define LOFF_A      (LOFF_H1 + 64 * SH1_STRIDE * 2)   // 83968
#define ABUF_BYTES  (64 * SA_STRIDE * 2)              // 5120
#define LOFF_BT     (LOFF_A + 2 * ABUF_BYTES)         // 94208
#define BTBUF_BYTES (512 * SBT_STRIDE * 2)            // 40960
#define LOFF_Q      (LOFF_BT + 2 * BTBUF_BYTES)       // 176128
#define SMEM_BYTES  (LOFF_Q + 64 * 4)                 // 176384

__global__ __launch_bounds__(256) void flat_critic_main(
    const float* __restrict__ adesc, const float* __restrict__ astd,
    const int* __restrict__ type_ids, const int* __restrict__ lengths,
    const float* __restrict__ b_desc, const float* __restrict__ b_std,
    const float* __restrict__ ctxp, const float* __restrict__ b2,
    const float* __restrict__ W3, const float* __restrict__ b3,
    const __bf16* __restrict__ wdT, const __bf16* __restrict__ wsT,
    const __bf16* __restrict__ w1eT, const __bf16* __restrict__ w2T,
    float* __restrict__ qout, float* __restrict__ embout) {

    extern __shared__ __align__(16) char smem[];
    __bf16* sEmb = (__bf16*)(smem + LOFF_EMB);   // 64 x 136
    __bf16* sH1  = (__bf16*)(smem + LOFF_H1);    // 64 x 520
    float*  qbuf = (float*)(smem + LOFF_Q);

    const int tid  = threadIdx.x;
    const int lane = tid & 31;
    const int w    = tid >> 5;
    const int b    = blockIdx.x >> 3;
    const int l0   = (blockIdx.x & 7) * 64;

    const int r0 = 16 * (w & 3);   // wave's M-tile rows (local)
    const int ng = w >> 2;         // wave's N-half

    if (tid < 64) qbuf[tid] = 0.0f;
    const int lenb = lengths[b];

    // ---- staging helpers (double-buffered) ----
    auto sAb  = [&](int bb) -> __bf16* { return (__bf16*)(smem + LOFF_A  + bb * ABUF_BYTES); };
    auto sBtb = [&](int bb) -> __bf16* { return (__bf16*)(smem + LOFF_BT + bb * BTBUF_BYTES); };

    // activations: 64 rows x 32 K fp32 -> bf16 via VALU (conversion required)
    auto stage_act = [&](const float* __restrict__ src_base, int K, int kc, int bb) {
        int row = tid >> 2, c0 = (tid & 3) * 8;
        const float* src = src_base + (size_t)(b * LL + l0 + row) * K + kc + c0;
        __bf16* dst = sAb(bb) + row * SA_STRIDE + c0;
        __builtin_prefetch(src + K, 0, 0);
#pragma unroll
        for (int i = 0; i < 8; ++i) dst[i] = (__bf16)src[i];
    };
    // weights 128 N rows x 32 K: async global->LDS (2 x b128 per thread)
    auto stage_w128 = [&](const __bf16* __restrict__ wbase, int K, int kc, int bb) {
        int n = tid >> 1, c0 = (tid & 1) * 16;
        unsigned lo = (unsigned)(LOFF_BT + bb * BTBUF_BYTES + (n * SBT_STRIDE + c0) * 2);
        unsigned go = (unsigned)((n * K + kc + c0) * 2);
        async_b128(lo,      wbase, go);
        async_b128(lo + 16, wbase, go + 16);
    };
    // weights 512 N rows x 32 K: async global->LDS (8 x b128 per thread)
    auto stage_w512 = [&](const __bf16* __restrict__ wbase, int K, int kc, int bb) {
        int n0 = tid * 2;
#pragma unroll
        for (int r = 0; r < 2; ++r) {
            int n = n0 + r;
            unsigned lo = (unsigned)(LOFF_BT + bb * BTBUF_BYTES + n * SBT_STRIDE * 2);
            unsigned go = (unsigned)((n * K + kc) * 2);
#pragma unroll
            for (int q4 = 0; q4 < 4; ++q4) async_b128(lo + q4 * 16, wbase, go + q4 * 16);
        }
    };

    // ================= Stage A: emb = select(type)(act @ W) ==========================
    v8f accD[4], accS[4];
#pragma unroll
    for (int t = 0; t < 4; ++t) { accD[t] = (v8f){}; accS[t] = (v8f){}; }

    // GEMM1: [64x768] @ [768x128], 24 chunks, software-pipelined
    stage_act(adesc, DDESC, 0, 0);
    stage_w128(wdT, DDESC, 0, 0);
    for (int i = 0; i < 24; ++i) {
        wait_async();
        __syncthreads();
        int bb = i & 1;
        if (i + 1 < 24) { stage_act(adesc, DDESC, (i + 1) * 32, bb ^ 1); stage_w128(wdT, DDESC, (i + 1) * 32, bb ^ 1); }
        v16bf af = load_a_frag(sAb(bb), SA_STRIDE, r0, 0, lane);
#pragma unroll
        for (int t = 0; t < 4; ++t) {
            v16bf bf_ = load_b_frag(sBtb(bb), SBT_STRIDE, 64 * ng + 16 * t, lane);
            accD[t] = WMMA_BF16(af, bf_, accD[t]);
        }
    }
    // GEMM2: [64x128] @ [128x128], 4 chunks
    stage_act(astd, DSTD, 0, 0);
    stage_w128(wsT, DSTD, 0, 0);
    for (int i = 0; i < 4; ++i) {
        wait_async();
        __syncthreads();
        int bb = i & 1;
        if (i + 1 < 4) { stage_act(astd, DSTD, (i + 1) * 32, bb ^ 1); stage_w128(wsT, DSTD, (i + 1) * 32, bb ^ 1); }
        v16bf af = load_a_frag(sAb(bb), SA_STRIDE, r0, 0, lane);
#pragma unroll
        for (int t = 0; t < 4; ++t) {
            v16bf bf_ = load_b_frag(sBtb(bb), SBT_STRIDE, 64 * ng + 16 * t, lane);
            accS[t] = WMMA_BF16(af, bf_, accS[t]);
        }
    }
    // epilogue A: bias, type select, length mask; write emb (f32 global + bf16 LDS)
#pragma unroll
    for (int t = 0; t < 4; ++t) {
        int col = 64 * ng + 16 * t + (lane & 15);
        float bd = b_desc[col], bs = b_std[col];
#pragma unroll
        for (int j = 0; j < 8; ++j) {
            int lrow   = r0 + j + ((lane & 16) ? 8 : 0);
            int grow   = l0 + lrow;
            int ty     = type_ids[b * LL + grow];
            bool valid = grow < lenb;
            float e = valid ? ((ty == 0) ? (accD[t][j] + bd) : (accS[t][j] + bs)) : 0.0f;
            embout[((size_t)(b * LL + grow)) * DACT + col] = e;
            sEmb[lrow * SEMB_STRIDE + col] = (__bf16)e;
        }
    }

    // ================= Stage B: h1 = relu(ctxp + emb @ W1e) ==========================
    v8f acc1[16];
#pragma unroll
    for (int t = 0; t < 16; ++t) acc1[t] = (v8f){};
    const int cbase = 256 * ng;
    stage_w512(w1eT, DACT, 0, 0);
    for (int i = 0; i < 4; ++i) {
        wait_async();
        __syncthreads();
        int bb = i & 1;
        if (i + 1 < 4) stage_w512(w1eT, DACT, (i + 1) * 32, bb ^ 1);
        v16bf af = load_a_frag(sEmb, SEMB_STRIDE, r0, i * 32, lane);
#pragma unroll
        for (int t = 0; t < 16; ++t) {
            v16bf bf_ = load_b_frag(sBtb(bb), SBT_STRIDE, cbase + 16 * t, lane);
            acc1[t] = WMMA_BF16(af, bf_, acc1[t]);
        }
    }
#pragma unroll
    for (int t = 0; t < 16; ++t) {
        int col = cbase + 16 * t + (lane & 15);
        float cv = ctxp[b * HH + col];
#pragma unroll
        for (int j = 0; j < 8; ++j) {
            int lrow = r0 + j + ((lane & 16) ? 8 : 0);
            float h = fmaxf(cv + acc1[t][j], 0.0f);
            sH1[lrow * SH1_STRIDE + col] = (__bf16)h;
        }
    }

    // ================= Stage C: h2 = relu(h1@W2 + b2); q = relu(h2@W3 + b3) ==========
    v8f acc2[16];
#pragma unroll
    for (int t = 0; t < 16; ++t) acc2[t] = (v8f){};
    stage_w512(w2T, HH, 0, 0);
    for (int i = 0; i < 16; ++i) {
        wait_async();
        __syncthreads();
        int bb = i & 1;
        if (i + 1 < 16) stage_w512(w2T, HH, (i + 1) * 32, bb ^ 1);
        v16bf af = load_a_frag(sH1, SH1_STRIDE, r0, i * 32, lane);
#pragma unroll
        for (int t = 0; t < 16; ++t) {
            v16bf bf_ = load_b_frag(sBtb(bb), SBT_STRIDE, cbase + 16 * t, lane);
            acc2[t] = WMMA_BF16(af, bf_, acc2[t]);
        }
    }
    // fused b2 + relu + dot with W3, cross-lane reduce, accumulate per-row in LDS
    float qp[8];
#pragma unroll
    for (int j = 0; j < 8; ++j) qp[j] = 0.0f;
#pragma unroll
    for (int t = 0; t < 16; ++t) {
        int col = cbase + 16 * t + (lane & 15);
        float w3 = W3[col], bb2 = b2[col];
#pragma unroll
        for (int j = 0; j < 8; ++j) {
            float h2 = fmaxf(acc2[t][j] + bb2, 0.0f);
            qp[j] += h2 * w3;
        }
    }
#pragma unroll
    for (int j = 0; j < 8; ++j) {
#pragma unroll
        for (int off = 1; off < 16; off <<= 1) qp[j] += __shfl_xor(qp[j], off, 32);
    }
    if ((lane & 15) == 0) {
        int rbase = r0 + ((lane & 16) ? 8 : 0);
#pragma unroll
        for (int j = 0; j < 8; ++j) atomicAdd(&qbuf[rbase + j], qp[j]);
    }
    __syncthreads();
    if (tid < 64) {
        float q = fmaxf(qbuf[tid] + b3[0], 0.0f);
        qout[(size_t)b * LL + l0 + tid] = q;
    }
}

// -------------------- host entry -----------------------------------------------------
extern "C" void kernel_launch(void* const* d_in, const int* in_sizes, int n_in,
                              void* d_out, int out_size, void* d_ws, size_t ws_size,
                              hipStream_t stream) {
    const float* instr  = (const float*)d_in[0];
    const float* state  = (const float*)d_in[1];
    const float* hidden = (const float*)d_in[2];
    const float* adesc  = (const float*)d_in[3];
    const float* astd   = (const float*)d_in[4];
    const int*   types  = (const int*)d_in[5];
    const int*   lens   = (const int*)d_in[6];
    const float* Wd     = (const float*)d_in[7];
    const float* bd     = (const float*)d_in[8];
    const float* Ws     = (const float*)d_in[9];
    const float* bs     = (const float*)d_in[10];
    const float* W1     = (const float*)d_in[11];
    const float* b1     = (const float*)d_in[12];
    const float* W2     = (const float*)d_in[13];
    const float* b2     = (const float*)d_in[14];
    const float* W3     = (const float*)d_in[15];
    const float* b3     = (const float*)d_in[16];

    char* ws = (char*)d_ws;
    __bf16* wdT  = (__bf16*)(ws + OFF_WDT);
    __bf16* wsT  = (__bf16*)(ws + OFF_WST);
    __bf16* w1eT = (__bf16*)(ws + OFF_W1ET);
    __bf16* w2T  = (__bf16*)(ws + OFF_W2T);
    float*  ctxp = (float*)(ws + OFF_CTXP);

    float* qout   = (float*)d_out;            // [B,L,1]
    float* embout = qout + BB * LL;           // [B,L,128]

    prep_weights<<<1728, 256, 0, stream>>>(Wd, Ws, W1, W2, wdT, wsT, w1eT, w2T);
    ctx_gemm<<<(BB * HH) / 256, 256, 0, stream>>>(instr, state, hidden, W1, b1, ctxp);

    static int smem_set = 0;
    if (!smem_set) {  // host-side attribute set; idempotent, not stream work
        hipFuncSetAttribute((const void*)flat_critic_main,
                            hipFuncAttributeMaxDynamicSharedMemorySize, SMEM_BYTES);
        smem_set = 1;
    }
    flat_critic_main<<<BB * (LL / 64), 256, SMEM_BYTES, stream>>>(
        adesc, astd, types, lens, bd, bs, ctxp, b2, W3, b3,
        wdT, wsT, w1eT, w2T, qout, embout);
}
